// LossComputation_58686433132721
// MI455X (gfx1250) — matrix operand order repaired and checked by reading, not compiled, not run on hardware
//
#include <hip/hip_runtime.h>

typedef __attribute__((ext_vector_type(16))) _Float16 v16h;
typedef __attribute__((ext_vector_type(8)))  _Float16 v8h;
typedef __attribute__((ext_vector_type(8)))  float    v8f;
typedef __attribute__((ext_vector_type(4)))  unsigned v4u;
typedef __attribute__((ext_vector_type(8)))  int      v8i;
typedef __attribute__((ext_vector_type(4)))  int      v4i;

#define MARGIN 0.3f
#define SMOOTH 0.1f
#define PD_EPS 1e-6f

#if defined(__HIP_DEVICE_COMPILE__) && __has_builtin(__builtin_amdgcn_tensor_load_to_lds)
#define HAVE_TDM 1
#else
#define HAVE_TDM 0
#endif

__device__ __forceinline__ unsigned wang_hash(unsigned s) {
  s = (s ^ 61u) ^ (s >> 16);
  s *= 9u;
  s ^= s >> 4;
  s *= 0x27d4eb2du;
  s ^= s >> 15;
  return s;
}
__device__ __forceinline__ float hash_u01(unsigned x) {
  return (float)(x >> 8) * (1.0f / 16777216.0f);  // [0,1)
}

// ---------------------------------------------------------------- fp32 -> f16
__global__ void k_cvt_f16(const float* __restrict__ src, _Float16* __restrict__ dst, long n) {
  long i = (long)blockIdx.x * blockDim.x + threadIdx.x;
  long stride = (long)gridDim.x * blockDim.x;
  for (; i < n; i += stride) dst[i] = (_Float16)src[i];
}

// ------------------------------------------------- row L2 normalize (fp32 out)
__global__ void k_rownorm_f32(const float* __restrict__ x, float* __restrict__ y, int D) {
  int row = blockIdx.x;
  const float* xr = x + (long)row * D;
  float* yr = y + (long)row * D;
  __shared__ float s[128];
  float acc = 0.f;
  for (int d = threadIdx.x; d < D; d += 128) { float v = xr[d]; acc += v * v; }
  s[threadIdx.x] = acc; __syncthreads();
  for (int o = 64; o > 0; o >>= 1) {
    if (threadIdx.x < o) s[threadIdx.x] += s[threadIdx.x + o];
    __syncthreads();
  }
  float inv = 1.0f / fmaxf(sqrtf(s[0]), 1e-12f);
  for (int d = threadIdx.x; d < D; d += 128) yr[d] = xr[d] * inv;
}

// -------------------------------------------------- row L2 normalize (f16 out)
__global__ void k_rownorm_f16(const float* __restrict__ x, _Float16* __restrict__ y, int D) {
  int row = blockIdx.x;
  const float* xr = x + (long)row * D;
  _Float16* yr = y + (long)row * D;
  __shared__ float s[128];
  float acc = 0.f;
  for (int d = threadIdx.x; d < D; d += 128) { float v = xr[d]; acc += v * v; }
  s[threadIdx.x] = acc; __syncthreads();
  for (int o = 64; o > 0; o >>= 1) {
    if (threadIdx.x < o) s[threadIdx.x] += s[threadIdx.x + o];
    __syncthreads();
  }
  float inv = 1.0f / fmaxf(sqrtf(s[0]), 1e-12f);
  for (int d = threadIdx.x; d < D; d += 128) yr[d] = (_Float16)(xr[d] * inv);
}

// ------------------------------------------- per-row argmax (first occurrence)
__global__ void k_rowargmax(const float* __restrict__ x, int* __restrict__ out, int C) {
  int row = blockIdx.x;
  const float* xr = x + (long)row * C;
  __shared__ float sv[256];
  __shared__ int   si[256];
  float bv = -3.4e38f; int bi = 0x7fffffff;
  for (int c = threadIdx.x; c < C; c += 256) {
    float v = xr[c];
    if (v > bv || (v == bv && c < bi)) { bv = v; bi = c; }
  }
  sv[threadIdx.x] = bv; si[threadIdx.x] = bi; __syncthreads();
  for (int o = 128; o > 0; o >>= 1) {
    if (threadIdx.x < o) {
      float ov = sv[threadIdx.x + o]; int oi = si[threadIdx.x + o];
      if (ov > sv[threadIdx.x] || (ov == sv[threadIdx.x] && oi < si[threadIdx.x])) {
        sv[threadIdx.x] = ov; si[threadIdx.x] = oi;
      }
    }
    __syncthreads();
  }
  if (threadIdx.x == 0) out[row] = si[0];
}

// ---------------------- triplet sampling: uniform-hash argmax over pos/neg set
__global__ void k_sample(const int* __restrict__ labels, int Bn, int Cm1, unsigned seed,
                         int* __restrict__ pos_idx, int* __restrict__ neg_idx) {
  int i = blockIdx.x;
  int li = labels[i]; li = min(max(li, 0), Cm1);
  __shared__ float spr[256]; __shared__ int spj[256];
  __shared__ float snr[256]; __shared__ int snj[256];
  float pr = -1.f; int pj = -1;
  float nr = -1.f; int nj = -1;
  for (int j = threadIdx.x; j < Bn; j += 256) {
    int lj = labels[j]; lj = min(max(lj, 0), Cm1);
    float r = hash_u01(wang_hash(seed ^ (unsigned)(i * Bn + j)));
    if (lj == li) {
      if (j != i && (r > pr || (r == pr && j < pj))) { pr = r; pj = j; }
    } else {
      if (r > nr || (r == nr && j < nj)) { nr = r; nj = j; }
    }
  }
  spr[threadIdx.x] = pr; spj[threadIdx.x] = pj;
  snr[threadIdx.x] = nr; snj[threadIdx.x] = nj;
  __syncthreads();
  for (int o = 128; o > 0; o >>= 1) {
    if (threadIdx.x < o) {
      float orp = spr[threadIdx.x + o]; int ojp = spj[threadIdx.x + o];
      if (orp > spr[threadIdx.x] || (orp == spr[threadIdx.x] && ojp < spj[threadIdx.x])) {
        spr[threadIdx.x] = orp; spj[threadIdx.x] = ojp;
      }
      float orn = snr[threadIdx.x + o]; int ojn = snj[threadIdx.x + o];
      if (orn > snr[threadIdx.x] || (orn == snr[threadIdx.x] && ojn < snj[threadIdx.x])) {
        snr[threadIdx.x] = orn; snj[threadIdx.x] = ojn;
      }
    }
    __syncthreads();
  }
  if (threadIdx.x == 0) {
    int p = spj[0];
    if (p < 0) p = (int)(wang_hash(seed ^ 0xABCD1234u ^ (unsigned)i) % (unsigned)Bn);
    int n = snj[0];
    if (n < 0) n = (int)(wang_hash(seed ^ 0x5678EF01u ^ (unsigned)i) % (unsigned)Bn);
    pos_idx[i] = p; neg_idx[i] = n;
  }
}

// ---------------------------- per-row triplet hinge: relu(|a-p+e|-|a-n+e|+m)
__global__ void k_trip(const float* __restrict__ f, const int* __restrict__ pos,
                       const int* __restrict__ neg, int D, float* __restrict__ out) {
  int row = blockIdx.x;
  const float* a = f + (long)row * D;
  const float* p = f + (long)pos[row] * D;
  const float* n = f + (long)neg[row] * D;
  __shared__ float sp[128], sn[128];
  float ap = 0.f, an = 0.f;
  for (int d = threadIdx.x; d < D; d += 128) {
    float av = a[d];
    float dp = av - p[d] + PD_EPS; ap += dp * dp;
    float dn = av - n[d] + PD_EPS; an += dn * dn;
  }
  sp[threadIdx.x] = ap; sn[threadIdx.x] = an; __syncthreads();
  for (int o = 64; o > 0; o >>= 1) {
    if (threadIdx.x < o) { sp[threadIdx.x] += sp[threadIdx.x + o]; sn[threadIdx.x] += sn[threadIdx.x + o]; }
    __syncthreads();
  }
  if (threadIdx.x == 0) out[row] = fmaxf(sqrtf(sp[0]) - sqrtf(sn[0]) + MARGIN, 0.f);
}

// ------------------------------------------------------------------ TDM issue
// 2D tile load: tileRows x tileK f16 elements, row stride rowStride elements,
// LDS rows padded +16B (pad_interval=32 DWORDs, pad_amount=4 DWORDs) so the
// staged row stride is tileK+8 halves (bank-conflict-free b128 reads).
#if HAVE_TDM
__device__ __forceinline__ void tdm_load_2d(unsigned ldsOff, const _Float16* gbase,
                                            long elemOff, int tileK, int tileRows,
                                            int tensorRows, int rowStride) {
  unsigned long long ga = (unsigned long long)(const void*)(gbase + elemOff);
  v4u g0;
  g0.x = 1u;                                              // count=1, user D#
  g0.y = ldsOff;                                          // lds_addr (bytes)
  g0.z = (unsigned)(ga & 0xffffffffull);                  // global_addr[31:0]
  g0.w = (unsigned)((ga >> 32) & 0x1ffffffull) | (2u << 30);  // addr[56:32] | type=2
  v8i g1;
  g1[0] = (int)((1u << 16) | (1u << 20) | (4u << 22) | (3u << 25));
  //            data_size=2B  pad_enable  interval=32DW  amount=4DW ; mask=0
  g1[1] = (int)((unsigned)tileK << 16);                   // tensor_dim0[15:0] @63:48
  g1[2] = (int)(((unsigned)tensorRows & 0xffffu) << 16);  // dim0[31:16]=0 | dim1[15:0]
  g1[3] = (int)(((unsigned)tileK << 16) |
                (((unsigned)tensorRows >> 16) & 0xffffu)); // dim1[31:16] | tile_dim0
  g1[4] = (int)((unsigned)tileRows & 0xffffu);            // tile_dim1 | tile_dim2=0
  g1[5] = (int)(unsigned)rowStride;                       // tensor_dim0_stride[31:0]
  g1[6] = 0;                                              // stride[47:32] | dim1_stride lo
  g1[7] = 0;
  v4i gz = {0, 0, 0, 0};
#if __clang_major__ >= 23
  v8i gz8 = {0, 0, 0, 0, 0, 0, 0, 0};
  __builtin_amdgcn_tensor_load_to_lds(g0, g1, gz, gz, gz8, 0);
#else
  __builtin_amdgcn_tensor_load_to_lds(g0, g1, gz, gz, 0);
#endif
}
#endif

// --------------------------------------------------------------- fused GEMM+CE
// block = 8 waves, tile 64 rows x 128 cols. K chunked by 64: A(64x64) and
// B(128x64) f16 tiles staged in LDS by the Tensor Data Mover (double-buffered,
// TENSORcnt), WMMA f32_16x16x32_f16 from LDS, softmax partials in epilogue.
union AFrag { v16h v; v8h h[2]; };

#define KT 64
#define LDA 72  /* KT + 8 halves: 144B row stride -> conflict-free b128 */

__global__ void __launch_bounds__(256)
k_gemm_ce(const _Float16* __restrict__ Fh, const _Float16* __restrict__ Wh,
          const float* __restrict__ bias, const int* __restrict__ labels,
          int Bn, int Dn, int Cn, int nchunk,
          float* __restrict__ pexp, float* __restrict__ plog, float* __restrict__ lablog) {
  __shared__ __align__(16) _Float16 sA[2][64 * LDA];
  __shared__ __align__(16) _Float16 sB[2][128 * LDA];

  int tid = threadIdx.x;
  int lane = tid & 31;
  int wave = tid >> 5;
  int wr = wave & 3, wc = wave >> 2;
  int r0Block = blockIdx.y * 64;
  int c0Block = blockIdx.x * 128;
  int r0 = r0Block + wr * 16;
  int c0 = c0Block + wc * 64;
  int hi = lane >> 4, ln = lane & 15;

  v8f zero = {0.f, 0.f, 0.f, 0.f, 0.f, 0.f, 0.f, 0.f};
  v8f acc[4];
#pragma unroll
  for (int nt = 0; nt < 4; ++nt) acc[nt] = zero;

  const int NC = Dn / KT;  // 384/64 = 6 K-chunks

#if HAVE_TDM
  if (wave == 0) {  // uniform per-wave branch; TDM ignores EXEC
    unsigned aOff0 = (unsigned)(unsigned long long)(const void*)&sA[0][0];
    unsigned bOff0 = (unsigned)(unsigned long long)(const void*)&sB[0][0];
    tdm_load_2d(aOff0, Fh, (long)r0Block * Dn, KT, 64, min(64, Bn - r0Block), Dn);
    tdm_load_2d(bOff0, Wh, (long)c0Block * Dn, KT, 128, min(128, Cn - c0Block), Dn);
  }
#endif

  for (int c = 0; c < NC; ++c) {
#if HAVE_TDM
    int cur = c & 1;
    if (wave == 0) {
      if (c + 1 < NC) {
        int nb = (c + 1) & 1;
        unsigned aOff = (unsigned)(unsigned long long)(const void*)&sA[nb][0];
        unsigned bOff = (unsigned)(unsigned long long)(const void*)&sB[nb][0];
        long kpos = (long)(c + 1) * KT;
        tdm_load_2d(aOff, Fh, (long)r0Block * Dn + kpos, KT, 64, min(64, Bn - r0Block), Dn);
        tdm_load_2d(bOff, Wh, (long)c0Block * Dn + kpos, KT, 128, min(128, Cn - c0Block), Dn);
        __builtin_amdgcn_s_wait_tensorcnt(2);  // current chunk (first 2) done
      } else {
        __builtin_amdgcn_s_wait_tensorcnt(0);
      }
    }
#else
    int cur = 0;
    {  // cooperative staging fallback (no TDM builtin on this toolchain)
      int k = c * KT;
      for (int idx = tid; idx < 64 * (KT / 8); idx += 256) {
        int row = idx / (KT / 8), cg = idx % (KT / 8);
        int gr = min(r0Block + row, Bn - 1);
        *(v8h*)&sA[0][row * LDA + cg * 8] = *(const v8h*)(Fh + (long)gr * Dn + k + cg * 8);
      }
      for (int idx = tid; idx < 128 * (KT / 8); idx += 256) {
        int row = idx / (KT / 8), cg = idx % (KT / 8);
        int gr = min(c0Block + row, Cn - 1);
        *(v8h*)&sB[0][row * LDA + cg * 8] = *(const v8h*)(Wh + (long)gr * Dn + k + cg * 8);
      }
    }
#endif
    __syncthreads();

    // Fragment reads from LDS tiles.
    // A (16-bit 16x32): lane holds M=ln; halves = K in [hi*16, hi*16+16).
    // B (16-bit 32x16): lane holds N=ln; halves = K in [hi*16, hi*16+16).
    const _Float16* Ap = &sA[cur][(wr * 16 + ln) * LDA] + hi * 16;
    const _Float16* Bq0 = &sB[cur][(wc * 64 + ln) * LDA] + hi * 16;
    const _Float16* Bq1 = &sB[cur][(wc * 64 + 16 + ln) * LDA] + hi * 16;
    const _Float16* Bq2 = &sB[cur][(wc * 64 + 32 + ln) * LDA] + hi * 16;
    const _Float16* Bq3 = &sB[cur][(wc * 64 + 48 + ln) * LDA] + hi * 16;
#pragma unroll
    for (int k0 = 0; k0 < KT; k0 += 32) {
      AFrag a;
      a.h[0] = *(const v8h*)(Ap + k0);
      a.h[1] = *(const v8h*)(Ap + k0 + 8);
      AFrag b0; b0.h[0] = *(const v8h*)(Bq0 + k0); b0.h[1] = *(const v8h*)(Bq0 + k0 + 8);
      AFrag b1; b1.h[0] = *(const v8h*)(Bq1 + k0); b1.h[1] = *(const v8h*)(Bq1 + k0 + 8);
      AFrag b2; b2.h[0] = *(const v8h*)(Bq2 + k0); b2.h[1] = *(const v8h*)(Bq2 + k0 + 8);
      AFrag b3; b3.h[0] = *(const v8h*)(Bq3 + k0); b3.h[1] = *(const v8h*)(Bq3 + k0 + 8);
      acc[0] = __builtin_amdgcn_wmma_f32_16x16x32_f16(false, a.v, false, b0.v, (short)0, acc[0], false, false);
      acc[1] = __builtin_amdgcn_wmma_f32_16x16x32_f16(false, a.v, false, b1.v, (short)0, acc[1], false, false);
      acc[2] = __builtin_amdgcn_wmma_f32_16x16x32_f16(false, a.v, false, b2.v, (short)0, acc[2], false, false);
      acc[3] = __builtin_amdgcn_wmma_f32_16x16x32_f16(false, a.v, false, b3.v, (short)0, acc[3], false, false);
    }
    __syncthreads();
  }

  // Epilogue. C/D layout: VGPR r of lane L holds row (r + hi*8), col ln.
  int rowBase = r0 + hi * 8;
  int lab[8];
#pragma unroll
  for (int r = 0; r < 8; ++r) {
    int row = rowBase + r;
    if (row < Bn) {
      int l = labels[row];
      lab[r] = min(max(l, 0), Cn - 1);
    } else {
      lab[r] = -1;
    }
  }
  float se[8], sl[8];
#pragma unroll
  for (int r = 0; r < 8; ++r) { se[r] = 0.f; sl[r] = 0.f; }

#pragma unroll
  for (int nt = 0; nt < 4; ++nt) {
    int n = c0 + nt * 16 + ln;
    bool valid = (n < Cn);
    float bn = valid ? bias[n] : 0.f;
#pragma unroll
    for (int r = 0; r < 8; ++r) {
      float v = acc[nt][r] + bn;
      if (valid) {
        se[r] += __expf(v);   // |logit| bounded (~1.5): no max-shift needed
        sl[r] += v;
        if (lab[r] == n) lablog[rowBase + r] = v;  // unique writer per row
      }
    }
  }

  // reduce across the 16-lane N group (xor masks stay inside the half-wave)
#pragma unroll
  for (int r = 0; r < 8; ++r) {
#pragma unroll
    for (int m = 1; m <= 8; m <<= 1) {
      se[r] += __shfl_xor(se[r], m, 32);
      sl[r] += __shfl_xor(sl[r], m, 32);
    }
  }
  if (ln == 0) {
    int chunk = blockIdx.x * 2 + wc;
#pragma unroll
    for (int r = 0; r < 8; ++r) {
      int row = rowBase + r;
      if (row < Bn) {
        pexp[(long)row * nchunk + chunk] = se[r];
        plog[(long)row * nchunk + chunk] = sl[r];
      }
    }
  }
}

// ------------------------------------------------- per-row smoothed CE finish
__global__ void k_ce_rows(const float* __restrict__ pexp, const float* __restrict__ plog,
                          const float* __restrict__ lablog, int Bn, int Cn, int nchunk,
                          float* __restrict__ cer) {
  int row = blockIdx.x * blockDim.x + threadIdx.x;
  if (row >= Bn) return;
  const float* pe = pexp + (long)row * nchunk;
  const float* pl = plog + (long)row * nchunk;
  float se = 0.f, sl = 0.f;
  for (int i = 0; i < nchunk; ++i) { se += pe[i]; sl += pl[i]; }
  float lse = logf(se);
  float nll = lse - lablog[row];
  float uni = lse - sl / (float)Cn;
  cer[row] = (1.0f - SMOOTH) * nll + SMOOTH * uni;
}

// ------------------------------------------------------------- final scalar
__global__ void k_final(const float* __restrict__ tm, const float* __restrict__ tt,
                        const float* __restrict__ ce, const float* __restrict__ moco,
                        int Bn, float* __restrict__ out) {
  __shared__ float s0[256], s1[256], s2[256];
  float a = 0.f, b = 0.f, c = 0.f;
  for (int i = threadIdx.x; i < Bn; i += 256) { a += tm[i]; b += tt[i]; c += ce[i]; }
  s0[threadIdx.x] = a; s1[threadIdx.x] = b; s2[threadIdx.x] = c; __syncthreads();
  for (int o = 128; o > 0; o >>= 1) {
    if (threadIdx.x < o) {
      s0[threadIdx.x] += s0[threadIdx.x + o];
      s1[threadIdx.x] += s1[threadIdx.x + o];
      s2[threadIdx.x] += s2[threadIdx.x + o];
    }
    __syncthreads();
  }
  if (threadIdx.x == 0) {
    float invB = 1.0f / (float)Bn;
    out[0] = moco[0] + 0.5f * ((s0[0] + s1[0]) * invB) + s2[0] * invB;
  }
}

extern "C" void kernel_launch(void* const* d_in, const int* in_sizes, int n_in,
                              void* d_out, int out_size, void* d_ws, size_t ws_size,
                              hipStream_t stream) {
  const float* F_mixed       = (const float*)d_in[0];
  const float* F_target      = (const float*)d_in[1];
  const float* F_source      = (const float*)d_in[2];
  const float* mixed_labels  = (const float*)d_in[3];
  const int*   pseudo_labels = (const int*)d_in[4];
  const int*   source_labels = (const int*)d_in[5];
  /* d_in[6] soft_probs: unused by the reference */
  const float* moco          = (const float*)d_in[7];
  const float* W             = (const float*)d_in[8];
  const float* bias          = (const float*)d_in[9];

  const int Bn = in_sizes[4];            // 4096
  const int Dn = in_sizes[0] / Bn;       // 384
  const int Cn = in_sizes[9];            // 12936
  const int NCB = (Cn + 127) / 128;      // col blocks of 128
  const int nchunk = NCB * 2;            // two 64-col wave groups per block

  char* ws = (char*)d_ws;
  size_t off = 0;
  auto wsalloc = [&](size_t bytes) -> void* {
    void* p = (void*)(ws + off);
    off = (off + bytes + 255) & ~(size_t)255;
    return p;
  };
  float*    Fm    = (float*)wsalloc((size_t)Bn * Dn * sizeof(float));
  float*    Ft    = (float*)wsalloc((size_t)Bn * Dn * sizeof(float));
  _Float16* FsH   = (_Float16*)wsalloc((size_t)Bn * Dn * sizeof(_Float16));
  _Float16* Wh    = (_Float16*)wsalloc((size_t)Cn * Dn * sizeof(_Float16));
  int*      mixA  = (int*)wsalloc((size_t)Bn * sizeof(int));
  int*      posm  = (int*)wsalloc((size_t)Bn * sizeof(int));
  int*      negm  = (int*)wsalloc((size_t)Bn * sizeof(int));
  int*      post  = (int*)wsalloc((size_t)Bn * sizeof(int));
  int*      negt  = (int*)wsalloc((size_t)Bn * sizeof(int));
  float*    tripm = (float*)wsalloc((size_t)Bn * sizeof(float));
  float*    tript = (float*)wsalloc((size_t)Bn * sizeof(float));
  float*    pexp  = (float*)wsalloc((size_t)Bn * nchunk * sizeof(float));
  float*    plog  = (float*)wsalloc((size_t)Bn * nchunk * sizeof(float));
  float*    labl  = (float*)wsalloc((size_t)Bn * sizeof(float));
  float*    cer   = (float*)wsalloc((size_t)Bn * sizeof(float));
  (void)ws_size; (void)n_in; (void)out_size;

  k_cvt_f16<<<1024, 256, 0, stream>>>(W, Wh, (long)Cn * Dn);
  k_rownorm_f32<<<Bn, 128, 0, stream>>>(F_mixed, Fm, Dn);
  k_rownorm_f32<<<Bn, 128, 0, stream>>>(F_target, Ft, Dn);
  k_rownorm_f16<<<Bn, 128, 0, stream>>>(F_source, FsH, Dn);
  k_rowargmax<<<Bn, 256, 0, stream>>>(mixed_labels, mixA, Cn);
  k_sample<<<Bn, 256, 0, stream>>>(mixA, Bn, Cn - 1, 0x01234567u, posm, negm);
  k_sample<<<Bn, 256, 0, stream>>>(pseudo_labels, Bn, Cn - 1, 0x89ABCDEFu, post, negt);
  k_trip<<<Bn, 128, 0, stream>>>(Fm, posm, negm, Dn, tripm);
  k_trip<<<Bn, 128, 0, stream>>>(Ft, post, negt, Dn, tript);

  dim3 g(NCB, (Bn + 63) / 64);
  k_gemm_ce<<<g, 256, 0, stream>>>(FsH, Wh, bias, source_labels,
                                   Bn, Dn, Cn, nchunk, pexp, plog, labl);
  k_ce_rows<<<(Bn + 255) / 256, 256, 0, stream>>>(pexp, plog, labl, Bn, Cn, nchunk, cer);
  k_final<<<1, 256, 0, stream>>>(tripm, tript, cer, moco, Bn, (float*)d_out);
}
